// DistancePredictor_74869869904130
// MI455X (gfx1250) — matrix operand order repaired and checked by reading, not compiled, not run on hardware
//
#include <hip/hip_runtime.h>
#include <hip/hip_bf16.h>

// ---------------------------------------------------------------------------
// DistancePredictor on MI455X (gfx1250, wave32, WMMA)
//   xi = x@Wi+bi ; xj = x@Wj+bj            [2,384,256]
//   out[b,i,j,:] = relu(xi[b,i]*xj[b,j]) @ Wo + bo     [2,384,384,10]
// Both GEMM stages run on v_wmma_f32_16x16x32_f16 (f16 in, f32 accumulate).
// ---------------------------------------------------------------------------

typedef _Float16 half_t;
typedef half_t v8h  __attribute__((ext_vector_type(8)));
typedef half_t v16h __attribute__((ext_vector_type(16)));
typedef float  v8f  __attribute__((ext_vector_type(8)));

#define BB   2
#define LL   384
#define DD   1280
#define HH   256
#define NB   10
#define NBP  16
#define ROWS (BB * LL)   // 768
#define NC   (2 * HH)    // 512 (xi | xj concatenated)

// ---------------------------------------------------------------------------
// Prep: transpose/convert weights to f16, K-contiguous per output column
//   WhT[n][k] = (n<256 ? Wi[k][n] : Wj[k][n-256])   -> [512][1280] f16
//   WoT[n][h] = (n<10 ? Wo[h][n] : 0)               -> [16][256]   f16
// ---------------------------------------------------------------------------
__global__ void prep_weights(const float* __restrict__ Wi,
                             const float* __restrict__ Wj,
                             const float* __restrict__ Wo,
                             half_t* __restrict__ WhT,
                             half_t* __restrict__ WoT) {
    int idx = blockIdx.x * 256 + threadIdx.x;
    if (idx < NC * DD) {
        int n = idx / DD, k = idx % DD;
        float v = (n < HH) ? Wi[k * HH + n] : Wj[k * HH + (n - HH)];
        WhT[idx] = (half_t)v;
    }
    if (idx < NBP * HH) {
        int n = idx / HH, h = idx % HH;
        float v = (n < NB) ? Wo[h * NB + n] : 0.0f;
        WoT[idx] = (half_t)v;
    }
}

// ---------------------------------------------------------------------------
// Stage 1: XIJ[m][n] = f16( (x @ [Wi|Wj])[m][n] + bias[n] )   m<768, n<512
// One wave per 16x16 tile, K=1280 in 40 WMMA steps.
// ---------------------------------------------------------------------------
__global__ void gemm_xij(const float* __restrict__ x,
                         const float* __restrict__ bi,
                         const float* __restrict__ bj,
                         const half_t* __restrict__ WhT,
                         half_t* __restrict__ XIJ) {
    int wave  = (blockIdx.x * blockDim.x + threadIdx.x) >> 5;
    int lane  = threadIdx.x & 31;
    int mtile = wave >> 5;          // 0..47  (32 n-tiles)
    int ntile = wave & 31;          // 0..31
    int hl    = lane >> 4;          // half of wave
    int l16   = lane & 15;

    int m = mtile * 16 + l16;       // A row for this lane
    int n = ntile * 16 + l16;       // B column for this lane
    const float*  xrow = x   + (size_t)m * DD;
    const half_t* wrow = WhT + (size_t)n * DD;

    v8f acc = {};
    for (int k0 = 0; k0 < DD; k0 += 32) {
        // A fragment: K = (e&8)*2 + hl*8 + (e&7) -> two 8-float runs, cvt f32->f16
        v8f xa = *(const v8f*)(xrow + k0 + hl * 8);
        v8f xb = *(const v8f*)(xrow + k0 + 16 + hl * 8);
        v16h a;
#pragma unroll
        for (int e = 0; e < 8; ++e) { a[e] = (half_t)xa[e]; a[e + 8] = (half_t)xb[e]; }
        // B fragment: K = (e&8)*2 + hl*8 + (e&7) -> two 8-half runs (16B each)
        v8h wb0 = *(const v8h*)(wrow + k0 + hl * 8);
        v8h wb1 = *(const v8h*)(wrow + k0 + 16 + hl * 8);
        v16h bfrag = __builtin_shufflevector(wb0, wb1, 0,1,2,3,4,5,6,7,8,9,10,11,12,13,14,15);
        acc = __builtin_amdgcn_wmma_f32_16x16x32_f16(false, a, false, bfrag,
                                                     (short)0, acc, false, false);
    }

    float bias = (n < HH) ? bi[n] : bj[n - HH];
#pragma unroll
    for (int r = 0; r < 8; ++r) {
        int mr = mtile * 16 + r + hl * 8;   // C/D row = vgpr + half*8
        XIJ[(size_t)mr * NC + n] = (half_t)(acc[r] + bias);
    }
}

// ---------------------------------------------------------------------------
// Stage 2: per (b,i): out[j,o] = relu(xi[i,:]*xj[j,:]) @ Wo + bo
// 1 block per (b,i); 8 waves x 3 j-tiles; K=256 in 8 WMMA steps.
// Wo B-fragments preloaded once per wave (8 x v16h = 64 VGPRs).
// ---------------------------------------------------------------------------
__global__ void pair_gemm(const half_t* __restrict__ XIJ,
                          const half_t* __restrict__ WoT,
                          const float* __restrict__ bo,
                          float* __restrict__ out) {
    int ri   = blockIdx.x;          // b*384 + i
    int bidx = ri / LL;
    int widx = threadIdx.x >> 5;    // wave in block (uniform per wave)
    int lane = threadIdx.x & 31;
    int hl   = lane >> 4;
    int l16  = lane & 15;

    // Preload all Wo fragments: B lane = column l16, K contiguous in WoT[l16][...]
    const half_t* worow = WoT + (size_t)l16 * HH;
    v16h bfrag[8];
#pragma unroll
    for (int kk = 0; kk < 8; ++kk)
        bfrag[kk] = __builtin_shufflevector(*(const v8h*)(worow + kk * 32 + hl * 8),
                                            *(const v8h*)(worow + kk * 32 + 16 + hl * 8),
                                            0,1,2,3,4,5,6,7,8,9,10,11,12,13,14,15);

    const half_t* xi = XIJ + (size_t)ri * NC;       // xi row (cols 0..255)
    float bov = (l16 < NB) ? bo[l16] : 0.0f;

    for (int jt = widx; jt < LL / 16; jt += 8) {
        int j = jt * 16 + l16;                      // A row for this lane
        const half_t* xj = XIJ + ((size_t)(bidx * LL + j)) * NC + HH;
        v8f acc = {};
#pragma unroll
        for (int kk = 0; kk < 8; ++kk) {
            int h0 = kk * 32;
            v8h xilo = *(const v8h*)(xi + h0 + hl * 8);
            v8h xihi = *(const v8h*)(xi + h0 + 16 + hl * 8);
            v8h xjlo = *(const v8h*)(xj + h0 + hl * 8);
            v8h xjhi = *(const v8h*)(xj + h0 + 16 + hl * 8);
            v8h plo = xilo * xjlo;                  // v_pk_mul_f16
            v8h phi = xihi * xjhi;
            v16h a;
            const half_t zl = (half_t)0;
#pragma unroll
            for (int e = 0; e < 8; ++e) {
                a[e]     = plo[e] > zl ? plo[e] : zl;   // relu, v_pk_max
                a[e + 8] = phi[e] > zl ? phi[e] : zl;
            }
            acc = __builtin_amdgcn_wmma_f32_16x16x32_f16(false, a, false, bfrag[kk],
                                                         (short)0, acc, false, false);
        }
        if (l16 < NB) {
#pragma unroll
            for (int r = 0; r < 8; ++r) {
                int jj = jt * 16 + r + hl * 8;
                out[(size_t)ri * (LL * NB) + (size_t)jj * NB + l16] = acc[r] + bov;
            }
        }
    }
}

// ---------------------------------------------------------------------------
extern "C" void kernel_launch(void* const* d_in, const int* in_sizes, int n_in,
                              void* d_out, int out_size, void* d_ws, size_t ws_size,
                              hipStream_t stream) {
    const float* x  = (const float*)d_in[0];
    const float* Wi = (const float*)d_in[1];
    const float* bi = (const float*)d_in[2];
    const float* Wj = (const float*)d_in[3];
    const float* bj = (const float*)d_in[4];
    const float* Wo = (const float*)d_in[5];
    const float* bo = (const float*)d_in[6];
    float* out = (float*)d_out;

    char* ws = (char*)d_ws;
    half_t* WhT = (half_t*)ws;                                   // 512*1280*2 = 1310720 B
    half_t* WoT = (half_t*)(ws + (size_t)NC * DD * 2);           // 16*256*2   =    8192 B
    half_t* XIJ = (half_t*)(ws + (size_t)NC * DD * 2 + (size_t)NBP * HH * 2); // 768*512*2

    prep_weights<<<(NC * DD + 255) / 256, 256, 0, stream>>>(Wi, Wj, Wo, WhT, WoT);
    gemm_xij<<<(ROWS / 16) * (NC / 16) * 32 / 256, 256, 0, stream>>>(x, bi, bj, WhT, XIJ);
    pair_gemm<<<ROWS, 256, 0, stream>>>(XIJ, WoT, bo, out);
}
